// TransformerBlock_9904194585048
// MI455X (gfx1250) — matrix operand order repaired and checked by reading, not compiled, not run on hardware
//
#include <hip/hip_runtime.h>
#include <hip/hip_bf16.h>

typedef __attribute__((ext_vector_type(16))) __bf16       v16bf;
typedef __attribute__((ext_vector_type(8)))  float        v8f;
typedef __attribute__((ext_vector_type(4)))  float        v4f;
typedef __attribute__((ext_vector_type(2)))  float        v2f;
typedef __attribute__((ext_vector_type(4)))  unsigned int v4u;
typedef __attribute__((ext_vector_type(4)))  unsigned int tdm_g0_t;
typedef __attribute__((ext_vector_type(8)))  int          tdm_g1_t;
typedef __attribute__((ext_vector_type(4)))  int          tdm_g2_t;

#define HW 9216
#define DIM 384
#define HEADS 8
#define HIDDEN 1021
#define PTOT (4 * HW)

#if __has_builtin(__builtin_amdgcn_tensor_load_to_lds) && \
    __has_builtin(__builtin_amdgcn_s_wait_tensorcnt)
#define HAVE_TDM 1
#else
#define HAVE_TDM 0
#endif

union FragCvt { v4u q[2]; v16bf v; };

// ---------------------------------------------------------------------------
// Weight convert: B[k][n] = w[n][k] (bf16, zero-padded to Kpad x Npad)
// ---------------------------------------------------------------------------
__global__ void wcvt_kernel(const float* __restrict__ w, __bf16* __restrict__ B,
                            int Kin, int Nin, int Kpad, int Npad) {
    int idx = blockIdx.x * 256 + threadIdx.x;
    if (idx >= Kpad * Npad) return;
    int k = idx / Npad, n = idx % Npad;
    float v = (k < Kin && n < Nin) ? w[(size_t)n * Kin + k] : 0.0f;
    B[idx] = (__bf16)v;
}

// ---------------------------------------------------------------------------
// LayerNorm over C=384 (per pixel), write bf16 pixel-major [P][384].
// One wave (32 lanes) per pixel; 8 pixels per 256-thread block.
// ---------------------------------------------------------------------------
__global__ __launch_bounds__(256) void ln_bf16_kernel(
    const float* __restrict__ x, const float* __restrict__ w,
    const float* __restrict__ bias, __bf16* __restrict__ out) {
    int wave = threadIdx.x >> 5, lane = threadIdx.x & 31;
    int p = blockIdx.x * 8 + wave;               // 0..36863
    int b = p / HW, hw = p % HW;
    const float* base = x + (size_t)b * DIM * HW + hw;
    float v[12];
    float s = 0.f, ss = 0.f;
    #pragma unroll
    for (int j = 0; j < 12; ++j) {
        float t = base[(size_t)(lane + 32 * j) * HW];
        v[j] = t; s += t; ss += t * t;
    }
    #pragma unroll
    for (int o = 16; o >= 1; o >>= 1) {
        s  += __shfl_xor(s,  o, 32);
        ss += __shfl_xor(ss, o, 32);
    }
    float mu  = s * (1.0f / DIM);
    float var = ss * (1.0f / DIM) - mu * mu;
    float inv = rsqrtf(var + 1e-5f);
    __bf16* orow = out + (size_t)p * DIM;
    #pragma unroll
    for (int j = 0; j < 12; ++j) {
        int c = lane + 32 * j;
        orow[c] = (__bf16)((v[j] - mu) * inv * w[c] + bias[c]);
    }
}

// ---------------------------------------------------------------------------
// bf16 GEMM: C[p][n] = sum_k A[p][k] * B[k][n]
//   A: [P][K] bf16 row-major   B: [K][Npad] bf16 row-major
//   out: f32 NCHW [4][Cout][HW]; optional residual add (same layout).
// Block: 256 thr = 8 waves; tile 128x64; wave tile 32x32 (2x2 wmma frags).
// Double-buffered pipeline: TDM (tensor_load_to_lds) stages A tile i+1 and
// cooperative dword loads stage transposed B tile i+1 while tile i runs on
// the WMMAs; consumption gated by s_wait_tensorcnt + one barrier per step.
// ---------------------------------------------------------------------------
template <bool HAS_RES>
__global__ __launch_bounds__(256) void gemm_bf16_kernel(
    const __bf16* __restrict__ A, const __bf16* __restrict__ B,
    float* __restrict__ out, const float* __restrict__ res,
    int K, int Npad, int Nreal, int Cout) {
    __shared__ __align__(16) __bf16 sa[2][128][32];   // A tiles, row-major [m][k]
    __shared__ __align__(16) __bf16 sbT[2][64][32];   // B tiles transposed [n][k]
    int t = threadIdx.x;
    int lane = t & 31, wv = t >> 5;
    int waveM = wv >> 1, waveN = wv & 1;              // 4 x 2 wave grid
    int half = lane >> 4, l15 = lane & 15;
    int nblocks = Npad / 64;
    int p0 = (blockIdx.x / nblocks) * 128;
    int n0 = (blockIdx.x % nblocks) * 64;

    v8f acc[2][2] = {};
    const unsigned int* Bu = (const unsigned int*)B;

#if HAVE_TDM
    // D# group1 (invariant): data_size=2B; tensor dims K x PTOT; tile 32x128;
    // dim0 stride = K elements.  (CDNA5 ISA)
    tdm_g1_t g1;
    g1[0] = (int)(1u << 16);                                        // data_size=1 -> 2B
    g1[1] = (int)(((unsigned)K & 0xFFFFu) << 16);                   // tensor_dim0 lo
    g1[2] = (int)((((unsigned)K >> 16) & 0xFFFFu) |
                  (((unsigned)PTOT & 0xFFFFu) << 16));              // dim0 hi | dim1 lo
    g1[3] = (int)((((unsigned)PTOT >> 16) & 0xFFFFu) | (32u << 16));// dim1 hi | tile_dim0=32
    g1[4] = 128;                                                    // tile_dim1=128, tile_dim2=0
    g1[5] = (int)(unsigned)K;                                       // tensor_dim0_stride lo
    g1[6] = 0;
    g1[7] = 0;
    tdm_g2_t gz = {0, 0, 0, 0};
    unsigned lds_a[2] = { (unsigned)(size_t)&sa[0][0][0],
                          (unsigned)(size_t)&sa[1][0][0] };
#endif

    // --- staging helpers -------------------------------------------------
    auto issueA = [&](int k0, int buf) {
#if HAVE_TDM
        if (wv == 0) {
            unsigned long long ga =
                (unsigned long long)(size_t)(A + (size_t)p0 * K + k0);
            tdm_g0_t g0;
            g0[0] = 1u;                                   // count=1 (valid user D#)
            g0[1] = lds_a[buf];                           // LDS dest
            g0[2] = (unsigned)ga;                         // global_addr lo
            g0[3] = (unsigned)((ga >> 32) & 0x1FFFFFFu) | (2u << 30); // hi | type=2
#if __clang_major__ >= 23
            __builtin_amdgcn_tensor_load_to_lds(g0, g1, gz, gz, (tdm_g1_t)0, 0);
#else
            __builtin_amdgcn_tensor_load_to_lds(g0, g1, gz, gz, 0);
#endif
        }
#else
        #pragma unroll
        for (int i = 0; i < 2; ++i) {
            int idx = t + i * 256;                 // 0..511 v4u slots
            int r = idx >> 2, c4 = idx & 3;
            ((v4u*)sa[buf])[idx] =
                *(const v4u*)(A + (size_t)(p0 + r) * K + k0 + c4 * 8);
        }
#endif
    };
    auto stageB = [&](int k0, int buf) {
        #pragma unroll
        for (int i = 0; i < 4; ++i) {
            int idx = t + i * 256;                 // 0..1023 dwords
            int r = idx >> 5, cu = idx & 31;       // k row, n pair
            unsigned d = Bu[((size_t)(k0 + r) * Npad + n0) / 2 + cu];
            int n_ = cu * 2;
            *(unsigned short*)&sbT[buf][n_][r]     = (unsigned short)(d & 0xFFFFu);
            *(unsigned short*)&sbT[buf][n_ + 1][r] = (unsigned short)(d >> 16);
        }
    };

    // --- pipelined main loop ---------------------------------------------
    int nk = K / 32;
    issueA(0, 0);
    stageB(0, 0);
#if HAVE_TDM
    if (wv == 0) __builtin_amdgcn_s_wait_tensorcnt(0);
#endif
    __syncthreads();

    for (int i = 0; i < nk; ++i) {
        int cur = i & 1, nxt = cur ^ 1;
        if (i + 1 < nk) {
            issueA((i + 1) * 32, nxt);             // DMA overlaps the WMMAs below
            stageB((i + 1) * 32, nxt);
            if (i + 2 < nk)                        // prefetch B tile i+2
                __builtin_prefetch(
                    (const char*)&Bu[((size_t)(i + 2) * 32 * Npad + n0) / 2], 0, 0);
        }

        v16bf af[2], bfr[2];
        #pragma unroll
        for (int mi = 0; mi < 2; ++mi) {           // two ds_load_b128 per frag
            int r  = waveM * 32 + mi * 16 + l15;
            int kb = half * 8;
            FragCvt fc;
            fc.q[0] = *(const v4u*)&sa[cur][r][kb];
            fc.q[1] = *(const v4u*)&sa[cur][r][16 + kb];
            af[mi] = fc.v;
        }
        #pragma unroll
        for (int ni = 0; ni < 2; ++ni) {           // two ds_load_b128 per frag
            int cc = waveN * 32 + ni * 16 + l15;
            int kb = half * 8;
            FragCvt fc;
            fc.q[0] = *(const v4u*)&sbT[cur][cc][kb];
            fc.q[1] = *(const v4u*)&sbT[cur][cc][16 + kb];
            bfr[ni] = fc.v;
        }
        #pragma unroll
        for (int mi = 0; mi < 2; ++mi)
            #pragma unroll
            for (int ni = 0; ni < 2; ++ni)
                acc[mi][ni] = __builtin_amdgcn_wmma_f32_16x16x32_bf16(
                    false, af[mi], false, bfr[ni], (short)0, acc[mi][ni], false, false);

        if (i + 1 < nk) {
#if HAVE_TDM
            if (wv == 0) __builtin_amdgcn_s_wait_tensorcnt(0);
#endif
            __syncthreads();                       // publish tile i+1; fence reuse
        }
    }

    // Epilogue: two float4 stores per fragment (8 consecutive hw positions)
    #pragma unroll
    for (int mi = 0; mi < 2; ++mi) {
        #pragma unroll
        for (int ni = 0; ni < 2; ++ni) {
            int n = n0 + waveN * 32 + ni * 16 + l15;
            if (n < Nreal) {
                int pbase = p0 + waveM * 32 + mi * 16 + 8 * half;
                int b = pbase / HW, hw = pbase % HW;
                size_t o = (size_t)b * Cout * HW + (size_t)n * HW + hw;
                v4f lo, hi;
                #pragma unroll
                for (int r = 0; r < 4; ++r) {
                    lo[r] = acc[mi][ni][r];
                    hi[r] = acc[mi][ni][r + 4];
                }
                if (HAS_RES) {
                    lo += *(const v4f*)&res[o];
                    hi += *(const v4f*)&res[o + 4];
                }
                *(v4f*)&out[o]     = lo;
                *(v4f*)&out[o + 4] = hi;
            }
        }
    }
}

// ---------------------------------------------------------------------------
// Depthwise 3x3 over qkv (NCHW 4x1152x96x96) + per-(b,c) 1/||.|| for q,k.
// One block per (b,c) plane; plane staged in LDS (36 KB).
// ---------------------------------------------------------------------------
__global__ __launch_bounds__(256) void dwconv_qkv_kernel(
    const float* __restrict__ in, const float* __restrict__ dww,
    float* __restrict__ outp, float* __restrict__ invn) {
    __shared__ float plane[HW];
    __shared__ float red[256];
    int bc = blockIdx.x;                          // b*1152 + c
    int c = bc % 1152, b = bc / 1152;
    const float* src = in + (size_t)bc * HW;
    for (int i = threadIdx.x; i < HW; i += 256) plane[i] = src[i];
    float wg[9];
    #pragma unroll
    for (int j = 0; j < 9; ++j) wg[j] = dww[c * 9 + j];
    __syncthreads();

    float ssq = 0.f;
    float* dst = outp + (size_t)bc * HW;
    for (int i = threadIdx.x; i < HW; i += 256) {
        int hh = i / 96, ww = i - hh * 96;
        float a = 0.f;
        #pragma unroll
        for (int kh = 0; kh < 3; ++kh) {
            int y = hh + kh - 1;
            if ((unsigned)y >= 96u) continue;
            #pragma unroll
            for (int kw = 0; kw < 3; ++kw) {
                int xw = ww + kw - 1;
                if ((unsigned)xw >= 96u) continue;
                a += wg[kh * 3 + kw] * plane[y * 96 + xw];
            }
        }
        dst[i] = a;
        ssq += a * a;
    }
    if (c < 768) {                                // q and k channels
        red[threadIdx.x] = ssq;
        __syncthreads();
        for (int o = 128; o >= 1; o >>= 1) {
            if (threadIdx.x < o) red[threadIdx.x] += red[threadIdx.x + o];
            __syncthreads();
        }
        if (threadIdx.x == 0)
            invn[b * 768 + c] = 1.0f / fmaxf(sqrtf(red[0]), 1e-12f);
    }
}

// ---------------------------------------------------------------------------
// Channel attention per (b,head): attn = softmax(qn.knT * temp); out = attn.v
// 9 waves; phase1: f32 WMMA q.kT with LDS 64-wide chunks; phase2: f32 WMMA.
// Output written bf16 pixel-major [P][384] (input to proj GEMM).
// ---------------------------------------------------------------------------
__global__ __launch_bounds__(288) void attn_kernel(
    const float* __restrict__ qkv, const float* __restrict__ invn,
    const float* __restrict__ temp, __bf16* __restrict__ A2) {
    __shared__ float qs[48][64];
    __shared__ float ks[48][64];
    __shared__ float at[48][49];
    int bh = blockIdx.x, b = bh >> 3, h = bh & 7;
    int t = threadIdx.x, lane = t & 31, wv = t >> 5;    // wv 0..8
    int half = lane >> 4, l15 = lane & 15;
    const float* qb = qkv + ((size_t)b * 1152 + h * 48) * HW;
    const float* kb = qkv + ((size_t)b * 1152 + 384 + h * 48) * HW;
    const float* vb = qkv + ((size_t)b * 1152 + 768 + h * 48) * HW;
    int ct = wv / 3, dt = wv % 3;                       // 3x3 tile pair per wave

    v8f acc = {};
    for (int n0 = 0; n0 < HW; n0 += 64) {
        for (int i = t; i < 48 * 64; i += 288) {
            int r = i >> 6, cc = i & 63;
            qs[r][cc] = qb[(size_t)r * HW + n0 + cc];
            ks[r][cc] = kb[(size_t)r * HW + n0 + cc];
        }
        __syncthreads();
        #pragma unroll
        for (int kk = 0; kk < 64; kk += 4) {
            v2f a, bb;
            int ar = ct * 16 + l15;
            a.x = qs[ar][kk + half * 2];
            a.y = qs[ar][kk + half * 2 + 1];
            int bc2 = dt * 16 + l15;
            bb.x = ks[bc2][kk + half * 2];
            bb.y = ks[bc2][kk + half * 2 + 1];
            acc = __builtin_amdgcn_wmma_f32_16x16x4_f32(
                false, a, false, bb, (short)0, acc, false, false);
        }
        __syncthreads();
    }

    // scale by l2-norm inverses * temperature, store to LDS
    {
        float tp = temp[h];
        int d = dt * 16 + l15;
        float ik = invn[b * 768 + 384 + h * 48 + d];
        #pragma unroll
        for (int r = 0; r < 8; ++r) {
            int cc = ct * 16 + r + 8 * half;
            float iq = invn[b * 768 + h * 48 + cc];
            at[cc][d] = acc[r] * iq * ik * tp;
        }
    }
    __syncthreads();
    if (t < 48) {                                       // row softmax
        float mx = -1e30f;
        for (int d = 0; d < 48; ++d) mx = fmaxf(mx, at[t][d]);
        float sm = 0.f;
        for (int d = 0; d < 48; ++d) { float e = __expf(at[t][d] - mx); at[t][d] = e; sm += e; }
        float rs = 1.0f / sm;
        for (int d = 0; d < 48; ++d) at[t][d] *= rs;
    }
    __syncthreads();

    // phase 2: out[c][n] = sum_d attn[c][d] * v[d][n]; 576 n-tiles x 3 c-tiles
    for (int task = wv; task < 1728; task += 9) {
        int ct2 = task % 3, nt = task / 3;
        int nn = nt * 16 + l15;
        v8f o = {};
        #pragma unroll
        for (int k0 = 0; k0 < 48; k0 += 4) {
            v2f a, bb;
            int ar = ct2 * 16 + l15;
            a.x = at[ar][k0 + half * 2];
            a.y = at[ar][k0 + half * 2 + 1];
            int dd = k0 + half * 2;
            bb.x = vb[(size_t)dd * HW + nn];
            bb.y = vb[(size_t)(dd + 1) * HW + nn];
            o = __builtin_amdgcn_wmma_f32_16x16x4_f32(
                false, a, false, bb, (short)0, o, false, false);
        }
        #pragma unroll
        for (int r = 0; r < 8; ++r) {
            int cc = ct2 * 16 + r + 8 * half;
            A2[((size_t)b * HW + nn) * DIM + h * 48 + cc] = (__bf16)o[r];
        }
    }
}

// ---------------------------------------------------------------------------
// FFN depthwise 3x3 + GELU gate: in NCHW [4][2042][HW] -> bf16 [P][1024]
// One block per (b, c<1021).  Row-band tiling: 6 bands of 16 rows; both
// channel planes' band+halo (2 x 18x96 f32 = 13.8 KB) staged in LDS, conv
// and gate computed in a single pass (no per-thread carry, no scratch).
// ---------------------------------------------------------------------------
__device__ __forceinline__ float gelu_tanh(float x) {
    float x3 = x * x * x;
    return 0.5f * x * (1.0f + tanhf(0.7978845608028654f * (x + 0.044715f * x3)));
}

__global__ __launch_bounds__(256) void dwconv_ffn_kernel(
    const float* __restrict__ in, const float* __restrict__ dww,
    __bf16* __restrict__ A4) {
    __shared__ float p1[18 * 96];
    __shared__ float p2[18 * 96];
    int blk = blockIdx.x;
    int c = blk % HIDDEN, b = blk / HIDDEN;
    const float* src1 = in + ((size_t)b * (2 * HIDDEN) + c) * HW;
    const float* src2 = src1 + (size_t)HIDDEN * HW;
    float wg1[9], wg2[9];
    #pragma unroll
    for (int j = 0; j < 9; ++j) {
        wg1[j] = dww[c * 9 + j];
        wg2[j] = dww[(c + HIDDEN) * 9 + j];
    }

    for (int band = 0; band < 6; ++band) {
        int rbase = band * 16 - 1;                 // global row of LDS row 0
        __syncthreads();                           // fence band reuse
        for (int i = threadIdx.x; i < 18 * 96; i += 256) {
            int rr = rbase + i / 96;
            int col = i % 96;
            float v1 = 0.f, v2 = 0.f;
            if ((unsigned)rr < 96u) {              // zero-pad top/bottom halo
                v1 = src1[rr * 96 + col];
                v2 = src2[rr * 96 + col];
            }
            p1[i] = v1;
            p2[i] = v2;
        }
        __syncthreads();
        for (int j = 0; j < 6; ++j) {              // 16*96/256 == 6
            int pi = threadIdx.x + j * 256;        // 0..1535 within band
            int hh = pi / 96, ww = pi - hh * 96;
            float a1 = 0.f, a2 = 0.f;
            #pragma unroll
            for (int kh = 0; kh < 3; ++kh) {
                int ly = (hh + kh) * 96;           // LDS row == global row - rbase
                #pragma unroll
                for (int kw = 0; kw < 3; ++kw) {
                    int xw = ww + kw - 1;
                    if ((unsigned)xw >= 96u) continue;
                    a1 += wg1[kh * 3 + kw] * p1[ly + xw];
                    a2 += wg2[kh * 3 + kw] * p2[ly + xw];
                }
            }
            int gp = band * 1536 + pi;             // pixel within plane
            A4[((size_t)b * HW + gp) * 1024 + c] = (__bf16)(gelu_tanh(a1) * a2);
        }
    }
}

// zero the 3 pad columns of A4 (K padded 1021 -> 1024)
__global__ void a4pad_kernel(__bf16* __restrict__ A4) {
    int idx = blockIdx.x * 256 + threadIdx.x;
    if (idx < PTOT * 3) {
        int p = idx / 3, j = idx % 3;
        A4[(size_t)p * 1024 + HIDDEN + j] = (__bf16)0.0f;
    }
}

// ---------------------------------------------------------------------------
extern "C" void kernel_launch(void* const* d_in, const int* in_sizes, int n_in,
                              void* d_out, int out_size, void* d_ws, size_t ws_size,
                              hipStream_t stream) {
    const float* x        = (const float*)d_in[0];
    const float* ln1w     = (const float*)d_in[1];
    const float* ln1b     = (const float*)d_in[2];
    const float* temp     = (const float*)d_in[3];
    const float* qkv_w    = (const float*)d_in[4];
    const float* qkv_dw   = (const float*)d_in[5];
    const float* proj_w   = (const float*)d_in[6];
    const float* ln2w     = (const float*)d_in[7];
    const float* ln2b     = (const float*)d_in[8];
    const float* ffn_in_w = (const float*)d_in[9];
    const float* ffn_dw   = (const float*)d_in[10];
    const float* ffn_outw = (const float*)d_in[11];
    float* out = (float*)d_out;

    char* wsp = (char*)d_ws;
    auto alloc = [&](size_t bytes) {
        char* p = wsp;
        wsp += (bytes + 255) & ~(size_t)255;
        return p;
    };
    __bf16* A1    = (__bf16*)alloc((size_t)PTOT * 384 * 2);
    __bf16* Bqkv  = (__bf16*)alloc((size_t)384 * 1152 * 2);
    __bf16* Bproj = (__bf16*)alloc((size_t)384 * 384 * 2);
    __bf16* Bff1  = (__bf16*)alloc((size_t)384 * 2048 * 2);
    __bf16* Bff2  = (__bf16*)alloc((size_t)1024 * 384 * 2);
    float*  qkvb  = (float*)alloc((size_t)4 * 1152 * HW * 4);
    float*  qkvd  = (float*)alloc((size_t)4 * 1152 * HW * 4);
    float*  invn  = (float*)alloc((size_t)4 * 768 * 4);
    __bf16* A2    = (__bf16*)alloc((size_t)PTOT * 384 * 2);
    float*  x1    = (float*)alloc((size_t)4 * 384 * HW * 4);
    __bf16* A3    = (__bf16*)alloc((size_t)PTOT * 384 * 2);
    float*  ffnh  = (float*)alloc((size_t)4 * 2042 * HW * 4);
    __bf16* A4    = (__bf16*)alloc((size_t)PTOT * 1024 * 2);

    // weights -> bf16 K-major
    wcvt_kernel<<<(384 * 1152 + 255) / 256, 256, 0, stream>>>(qkv_w,    Bqkv,  384, 1152, 384, 1152);
    wcvt_kernel<<<(384 * 384  + 255) / 256, 256, 0, stream>>>(proj_w,   Bproj, 384, 384,  384, 384);
    wcvt_kernel<<<(384 * 2048 + 255) / 256, 256, 0, stream>>>(ffn_in_w, Bff1,  384, 2042, 384, 2048);
    wcvt_kernel<<<(1024 * 384 + 255) / 256, 256, 0, stream>>>(ffn_outw, Bff2, 1021, 384, 1024, 384);

    // LN1 -> A1
    ln_bf16_kernel<<<PTOT / 8, 256, 0, stream>>>(x, ln1w, ln1b, A1);
    // qkv = A1 @ Bqkv -> NCHW f32
    gemm_bf16_kernel<false><<<(PTOT / 128) * (1152 / 64), 256, 0, stream>>>(
        A1, Bqkv, qkvb, nullptr, 384, 1152, 1152, 1152);
    // depthwise + q/k inverse norms
    dwconv_qkv_kernel<<<4 * 1152, 256, 0, stream>>>(qkvb, qkv_dw, qkvd, invn);
    // channel attention -> A2 (bf16 pixel-major)
    attn_kernel<<<32, 288, 0, stream>>>(qkvd, invn, temp, A2);
    // x1 = x + A2 @ Bproj
    gemm_bf16_kernel<true><<<(PTOT / 128) * (384 / 64), 256, 0, stream>>>(
        A2, Bproj, x1, x, 384, 384, 384, 384);
    // LN2 -> A3
    ln_bf16_kernel<<<PTOT / 8, 256, 0, stream>>>(x1, ln2w, ln2b, A3);
    // ffn hidden = A3 @ Bff1 -> NCHW f32 (2042 real of 2048)
    gemm_bf16_kernel<false><<<(PTOT / 128) * (2048 / 64), 256, 0, stream>>>(
        A3, Bff1, ffnh, nullptr, 384, 2048, 2042, 2042);
    // dw conv + gelu gate -> A4
    dwconv_ffn_kernel<<<4 * HIDDEN, 256, 0, stream>>>(ffnh, ffn_dw, A4);
    a4pad_kernel<<<(PTOT * 3 + 255) / 256, 256, 0, stream>>>(A4);
    // out = x1 + A4 @ Bff2
    gemm_bf16_kernel<true><<<(PTOT / 128) * (384 / 64), 256, 0, stream>>>(
        A4, Bff2, out, x1, 1024, 384, 384, 384);
}
